// DeepAR_77876347011377
// MI455X (gfx1250) — matrix operand order repaired and checked by reading, not compiled, not run on hardware
//
#include <hip/hip_runtime.h>
#include <hip/hip_bf16.h>
#include <math.h>

#define S_SAMPLES 8192
#define HIDDEN    512
#define EMB       40
#define TIME_FEAT 4
#define HIST      336
#define PRED      48
#define INDIM     41          // C_IN + EMB
#define G3        (3*HIDDEN)  // 1536
#define BLK_ROWS  32          // rows staged per block / computed per wave

typedef __attribute__((ext_vector_type(16))) __bf16 v16bf;
typedef __attribute__((ext_vector_type(8)))  float  v8f;

// ---- fast activations: single v_rcp_f32 instead of IEEE div sequence ----
__device__ __forceinline__ float fast_sigmoid(float x){
  return __builtin_amdgcn_rcpf(1.f + __expf(-x));
}
__device__ __forceinline__ float fast_tanh(float x){
  // tanh(x) = 2/(1+e^(-2x)) - 1
  return fmaf(2.f, __builtin_amdgcn_rcpf(1.f + __expf(-2.f*x)), -1.f);
}
__device__ __forceinline__ float softplusf_(float x){ return x>20.f ? x : log1pf(__expf(x)); }

// ---- counter-based RNG (PCG-style) ----
__device__ __forceinline__ unsigned pcg_next(unsigned &st){
  st = st*747796405u + 2891336453u;
  unsigned w = ((st >> ((st>>28)+4u)) ^ st) * 277803737u;
  return (w>>22) ^ w;
}
__device__ __forceinline__ float uni01(unsigned &st){
  return ((pcg_next(st)>>8)+1u) * (1.f/16777217.f);
}

// =====================================================================
// P0: scale = clip(mean|x|), enc_in = [x/scale , x_mark@W_emb.T+b],
//     emb_y = y_mark@W_emb.T+b.   Single block.
// =====================================================================
__global__ void preprocess(const float* __restrict__ x,
                           const float* __restrict__ x_mark,
                           const float* __restrict__ y_mark,
                           const float* __restrict__ W_emb,
                           const float* __restrict__ b_emb,
                           float* __restrict__ scale_p,
                           float* __restrict__ enc_in,
                           float* __restrict__ emb_y)
{
  __shared__ float red[512];
  __shared__ float ssc;
  const int tid = threadIdx.x;
  float acc = 0.f;
  for (int t = tid; t < HIST; t += 512) acc += fabsf(x[t]);
  red[tid] = acc; __syncthreads();
  for (int s = 256; s > 0; s >>= 1){ if (tid < s) red[tid] += red[tid+s]; __syncthreads(); }
  if (tid == 0){ float sc = fmaxf(red[0]/(float)HIST, 1e-5f); ssc = sc; scale_p[0] = sc; }
  __syncthreads();
  const float sc = ssc;

  for (int idx = tid; idx < HIST*INDIM; idx += 512){
    int t = idx / INDIM, i = idx % INDIM;
    float v;
    if (i == 0) v = x[t] / sc;
    else {
      int e = i - 1; float a = b_emb[e];
      #pragma unroll
      for (int f = 0; f < TIME_FEAT; ++f) a = fmaf(x_mark[t*TIME_FEAT+f], W_emb[e*TIME_FEAT+f], a);
      v = a;
    }
    enc_in[idx] = v;
  }
  for (int idx = tid; idx < PRED*EMB; idx += 512){
    int k = idx / EMB, e = idx % EMB;
    float a = b_emb[e];
    #pragma unroll
    for (int f = 0; f < TIME_FEAT; ++f) a = fmaf(y_mark[k*TIME_FEAT+f], W_emb[e*TIME_FEAT+f], a);
    emb_y[idx] = a;
  }
}

// =====================================================================
// Encoder GRU: batch 1, 336 sequential steps, 1 workgroup of 512 threads
// =====================================================================
__global__ void encoder_gru(const float* __restrict__ enc_in,
                            const float* __restrict__ W_ih,
                            const float* __restrict__ W_hh,
                            const float* __restrict__ b_ih,
                            const float* __restrict__ b_hh,
                            float* __restrict__ hT)
{
  __shared__ float hbuf[2][HIDDEN];
  const int j = threadIdx.x;
  hbuf[0][j] = 0.f; hbuf[1][j] = 0.f;
  __syncthreads();
  int cur = 0;
  const float* wir = W_ih + (size_t)j        * INDIM;
  const float* wiz = W_ih + (size_t)(j+512)  * INDIM;
  const float* win = W_ih + (size_t)(j+1024) * INDIM;
  const float* whr = W_hh + (size_t)j        * HIDDEN;
  const float* whz = W_hh + (size_t)(j+512)  * HIDDEN;
  const float* whn = W_hh + (size_t)(j+1024) * HIDDEN;
  for (int t = 0; t < HIST; ++t){
    const float* xt = enc_in + t*INDIM;
    float gir = b_ih[j], giz = b_ih[j+512], gin = b_ih[j+1024];
    for (int i = 0; i < INDIM; ++i){
      float xv = xt[i];
      gir = fmaf(xv, wir[i], gir); giz = fmaf(xv, wiz[i], giz); gin = fmaf(xv, win[i], gin);
    }
    float ghr = b_hh[j], ghz = b_hh[j+512], ghn = b_hh[j+1024];
    const float* hc = hbuf[cur];
    for (int i = 0; i < HIDDEN; ++i){
      float hv = hc[i];
      ghr = fmaf(hv, whr[i], ghr); ghz = fmaf(hv, whz[i], ghz); ghn = fmaf(hv, whn[i], ghn);
    }
    float r = fast_sigmoid(gir + ghr);
    float z = fast_sigmoid(giz + ghz);
    float n = fast_tanh(gin + r*ghn);
    float hv = (1.f - z)*n + z*hc[j];
    hbuf[cur^1][j] = hv;
    __syncthreads();
    cur ^= 1;
  }
  hT[j] = hbuf[cur][j];
}

// =====================================================================
// Weight prep: W_hh -> bf16 (L2-resident B operand), w0 = W_ih[:,0]
// =====================================================================
__global__ void prep_weights(const float* __restrict__ W_hh,
                             const float* __restrict__ W_ih,
                             __bf16* __restrict__ Wbf,
                             float* __restrict__ w0)
{
  int idx = blockIdx.x*blockDim.x + threadIdx.x;
  const int total = G3*HIDDEN;
  for (int i = idx; i < total; i += gridDim.x*blockDim.x) Wbf[i] = (__bf16)W_hh[i];
  if (idx < G3) w0[idx] = W_ih[(size_t)idx*INDIM];
}

// Ck[k][n] = b_ih[n] + sum_j emb_y[k][j] * W_ih[n][1+j]  (rank-1 residual)
__global__ void prep_ck(const float* __restrict__ emb_y,
                        const float* __restrict__ W_ih,
                        const float* __restrict__ b_ih,
                        float* __restrict__ Ck)
{
  int idx = blockIdx.x*blockDim.x + threadIdx.x;
  if (idx >= PRED*G3) return;
  int k = idx / G3, n = idx % G3;
  float acc = b_ih[n];
  const float* ey = emb_y + k*EMB;
  const float* w  = W_ih + (size_t)n*INDIM + 1;
  #pragma unroll 8
  for (int j = 0; j < EMB; ++j) acc = fmaf(ey[j], w[j], acc);
  Ck[idx] = acc;
}

// Broadcast h_T to all samples (f32 + bf16), tgt0 = raw last x value
__global__ void init_state(const float* __restrict__ hT,
                           const float* __restrict__ x,
                           float* __restrict__ h,
                           __bf16* __restrict__ hbf,
                           float* __restrict__ tgt)
{
  int idx = blockIdx.x*blockDim.x + threadIdx.x;
  if (idx < S_SAMPLES*HIDDEN){
    float v = hT[idx & (HIDDEN-1)];
    h[idx] = v; hbf[idx] = (__bf16)v;
  }
  if (idx < S_SAMPLES) tgt[idx] = x[HIST-1];
}

// =====================================================================
// Decoder GRU step via bf16 WMMA.
// Block = 8 waves = 256 threads. Block tile: 32 rows x 128 hidden cols.
// The 32x512 bf16 A tile (32KB) is cooperatively staged into LDS once,
// then every wave computes a 32x16 output tile: 2 row sub-tiles x 3 gates
// = 6 f32 accumulators, 6 v_wmma_f32_16x16x32_bf16 per 32-wide K chunk
// (16 chunks). A comes from LDS (ds_load_b128), B (W_hh bf16, 1.5MB)
// stays L2-resident. Input GEMM folded to rank-1: gi = tgt*w0[n]+Ck[n].
// =====================================================================
__global__ void gru_gates_wmma(const __bf16* __restrict__ hbf_cur,
                               const float*  __restrict__ h_cur,
                               float*        __restrict__ h_next,
                               __bf16*       __restrict__ hbf_next,
                               const __bf16* __restrict__ Wbf,   // [1536][512] bf16
                               const float*  __restrict__ b_hh,  // [1536]
                               const float*  __restrict__ w0,    // [1536] = W_ih[:,0]
                               const float*  __restrict__ Ck,    // [48][1536]
                               const float*  __restrict__ tgt,   // [8192]
                               int step)
{
  __shared__ alignas(16) __bf16 Ash[BLK_ROWS * HIDDEN];   // 32 KB

  const int lane = threadIdx.x & 31;
  const int wave = threadIdx.x >> 5;
  const int rowg = blockIdx.x >> 2;              // 256 row groups of 32
  const int colg = blockIdx.x & 3;               // 4 col groups of 128
  const int row0 = rowg * BLK_ROWS;
  const int col0 = colg * 128 + wave * 16;

  // ---- cooperative stage: 32 rows x 512 cols bf16 = 2048 x 16B chunks ----
  {
    const int t = threadIdx.x;
    #pragma unroll
    for (int i = 0; i < 8; ++i){
      int chunk = t + i*256;                     // 0..2047
      int r  = chunk >> 6;                       // 64 chunks per row
      int cc = (chunk & 63) * 8;
      *reinterpret_cast<uint4*>(&Ash[r*HIDDEN + cc]) =
        *reinterpret_cast<const uint4*>(hbf_cur + (size_t)(row0 + r)*HIDDEN + cc);
    }
  }
  __syncthreads();

  // A fragment (16x32 bf16): lane<16 -> row=lane,    K in {0..15}
  //                          lane>=16 -> row=lane-16, K in {16..31}
  const int ar0   = (lane & 15);                 // row sub-tile 0
  const int ar1   = 16 + (lane & 15);            // row sub-tile 1
  const int kselA = (lane >> 4) * 16;
  // B fragment (32x16 bf16): col = lane&15, K contiguous 16, base 0 or 16
  const int bn    = col0 + (lane & 15);
  const int kselB = (lane >> 4) * 16;

  const __bf16* BrowR = Wbf + (size_t)(bn)        * HIDDEN;
  const __bf16* BrowZ = Wbf + (size_t)(bn + 512)  * HIDDEN;
  const __bf16* BrowN = Wbf + (size_t)(bn + 1024) * HIDDEN;

  v8f aR0 = {}, aZ0 = {}, aN0 = {};
  v8f aR1 = {}, aZ1 = {}, aN1 = {};

  for (int k0 = 0; k0 < HIDDEN; k0 += 32){
    v16bf a0, a1, br, bz, bn_;
    #pragma unroll
    for (int i = 0; i < 16; ++i){
      a0[i]   = Ash[ar0*HIDDEN + k0 + kselA + i];
      a1[i]   = Ash[ar1*HIDDEN + k0 + kselA + i];
    }
    #pragma unroll
    for (int i = 0; i < 16; ++i){
      br[i]  = BrowR[k0 + kselB + i];
      bz[i]  = BrowZ[k0 + kselB + i];
      bn_[i] = BrowN[k0 + kselB + i];
    }
    aR0 = __builtin_amdgcn_wmma_f32_16x16x32_bf16(false, a0, false, br,  (short)0, aR0, false, false);
    aR1 = __builtin_amdgcn_wmma_f32_16x16x32_bf16(false, a1, false, br,  (short)0, aR1, false, false);
    aZ0 = __builtin_amdgcn_wmma_f32_16x16x32_bf16(false, a0, false, bz,  (short)0, aZ0, false, false);
    aZ1 = __builtin_amdgcn_wmma_f32_16x16x32_bf16(false, a1, false, bz,  (short)0, aZ1, false, false);
    aN0 = __builtin_amdgcn_wmma_f32_16x16x32_bf16(false, a0, false, bn_, (short)0, aN0, false, false);
    aN1 = __builtin_amdgcn_wmma_f32_16x16x32_bf16(false, a1, false, bn_, (short)0, aN1, false, false);
  }

  // Epilogue: C/D layout -> element r at M = r + (lane>>4)*8, N = lane&15
  const int n = col0 + (lane & 15);
  const float bhr = b_hh[n], bhz = b_hh[n+512], bhn = b_hh[n+1024];
  const float w0r = w0[n],  w0z = w0[n+512],  w0n = w0[n+1024];
  const float* ck = Ck + (size_t)step * G3;
  const float ckr = ck[n], ckz = ck[n+512], ckn = ck[n+1024];

  #pragma unroll
  for (int half = 0; half < 2; ++half){
    const v8f& hR = half ? aR1 : aR0;
    const v8f& hZ = half ? aZ1 : aZ0;
    const v8f& hN = half ? aN1 : aN0;
    #pragma unroll
    for (int r = 0; r < 8; ++r){
      int m = row0 + half*16 + r + (lane >> 4) * 8;
      float t  = tgt[m];
      float hr = hR[r] + bhr;
      float hz = hZ[r] + bhz;
      float hn = hN[r] + bhn;
      float rg = fast_sigmoid(fmaf(t, w0r, ckr) + hr);
      float zg = fast_sigmoid(fmaf(t, w0z, ckz) + hz);
      float ng = fast_tanh(fmaf(t, w0n, ckn) + rg * hn);
      float ho = h_cur[(size_t)m*HIDDEN + n];
      float hv = (1.f - zg)*ng + zg*ho;
      h_next[(size_t)m*HIDDEN + n]  = hv;
      hbf_next[(size_t)m*HIDDEN + n] = (__bf16)hv;
    }
  }
}

// =====================================================================
// Per-step heads: df/loc/sc projections (float4 dots) + Student-t sample
// =====================================================================
__global__ void proj_sample(const float* __restrict__ h,
                            const float* __restrict__ Wdf, const float* __restrict__ bdf,
                            const float* __restrict__ Wloc, const float* __restrict__ bloc,
                            const float* __restrict__ Wsc, const float* __restrict__ bsc,
                            const float* __restrict__ scale_p,
                            float* __restrict__ tgt,
                            float* __restrict__ out, int step)
{
  int s = blockIdx.x*blockDim.x + threadIdx.x;
  if (s >= S_SAMPLES) return;
  const float4* h4  = reinterpret_cast<const float4*>(h + (size_t)s*HIDDEN);
  const float4* wd4 = reinterpret_cast<const float4*>(Wdf);
  const float4* wl4 = reinterpret_cast<const float4*>(Wloc);
  const float4* ws4 = reinterpret_cast<const float4*>(Wsc);
  float adf = 0.f, aloc = 0.f, asc = 0.f;
  #pragma unroll 4
  for (int j = 0; j < HIDDEN/4; ++j){
    float4 hv = h4[j], wd = wd4[j], wl = wl4[j], wsv = ws4[j];
    adf  = fmaf(hv.x, wd.x,  fmaf(hv.y, wd.y,  fmaf(hv.z, wd.z,  fmaf(hv.w, wd.w,  adf))));
    aloc = fmaf(hv.x, wl.x,  fmaf(hv.y, wl.y,  fmaf(hv.z, wl.z,  fmaf(hv.w, wl.w,  aloc))));
    asc  = fmaf(hv.x, wsv.x, fmaf(hv.y, wsv.y, fmaf(hv.z, wsv.z, fmaf(hv.w, wsv.w, asc))));
  }
  float df  = 2.f + softplusf_(adf + bdf[0]);
  float loc = aloc + bloc[0];
  float sc  = softplusf_(asc + bsc[0]);

  unsigned st = (unsigned)s * 2654435761u ^ (unsigned)(step*0x9E3779B9u) ^ 42u;
  pcg_next(st);
  // standard normal (Box-Muller)
  float u1 = uni01(st), u2 = uni01(st);
  float n0 = sqrtf(-2.f*__logf(u1)) * __cosf(6.2831853f*u2);
  // chi2(df) = 2*Gamma(df/2) via Marsaglia-Tsang (bounded rejection)
  float d = 0.5f*df - (1.f/3.f);
  float c = rsqrtf(9.f*d);
  float g = d;
  for (int it = 0; it < 16; ++it){
    float xz = sqrtf(-2.f*__logf(uni01(st))) * __cosf(6.2831853f*uni01(st));
    float v = 1.f + c*xz; v = v*v*v;
    if (v <= 0.f) continue;
    float u = uni01(st);
    if (__logf(u) < 0.5f*xz*xz + d - d*v + d*__logf(v)){ g = d*v; break; }
  }
  float chi2 = 2.f*g;
  float eps = n0 * rsqrtf(chi2/df);
  float smp = (loc + sc*eps) * scale_p[0];
  out[(size_t)s*PRED + step] = smp;
  tgt[s] = smp;
}

// =====================================================================
extern "C" void kernel_launch(void* const* d_in, const int* in_sizes, int n_in,
                              void* d_out, int out_size, void* d_ws, size_t ws_size,
                              hipStream_t stream)
{
  const float* x      = (const float*)d_in[0];
  const float* x_mark = (const float*)d_in[1];
  const float* y_mark = (const float*)d_in[2];
  // d_in[3]: mode (int scalar) — unused
  const float* W_emb  = (const float*)d_in[4];
  const float* b_emb  = (const float*)d_in[5];
  const float* W_ih   = (const float*)d_in[6];
  const float* W_hh   = (const float*)d_in[7];
  const float* b_ih   = (const float*)d_in[8];
  const float* b_hh   = (const float*)d_in[9];
  const float* W_df   = (const float*)d_in[10];
  const float* b_df   = (const float*)d_in[11];
  const float* W_loc  = (const float*)d_in[12];
  const float* b_loc  = (const float*)d_in[13];
  const float* W_sc   = (const float*)d_in[14];
  const float* b_sc   = (const float*)d_in[15];
  float* out = (float*)d_out;

  char* base = (char*)d_ws;
  size_t cur = 0;
  auto alloc = [&](size_t bytes) -> void* {
    cur = (cur + 255) & ~(size_t)255;
    void* p = base + cur;
    cur += bytes;
    return p;
  };
  float*  scale_p = (float*)alloc(sizeof(float));
  float*  hT      = (float*)alloc(HIDDEN*sizeof(float));
  float*  enc_in  = (float*)alloc((size_t)HIST*INDIM*sizeof(float));
  float*  emb_y   = (float*)alloc((size_t)PRED*EMB*sizeof(float));
  float*  Ck      = (float*)alloc((size_t)PRED*G3*sizeof(float));
  float*  w0      = (float*)alloc((size_t)G3*sizeof(float));
  float*  tgt     = (float*)alloc((size_t)S_SAMPLES*sizeof(float));
  __bf16* Wbf     = (__bf16*)alloc((size_t)G3*HIDDEN*sizeof(__bf16));
  float*  hA      = (float*)alloc((size_t)S_SAMPLES*HIDDEN*sizeof(float));
  float*  hB      = (float*)alloc((size_t)S_SAMPLES*HIDDEN*sizeof(float));
  __bf16* hbfA    = (__bf16*)alloc((size_t)S_SAMPLES*HIDDEN*sizeof(__bf16));
  __bf16* hbfB    = (__bf16*)alloc((size_t)S_SAMPLES*HIDDEN*sizeof(__bf16));

  preprocess<<<1, 512, 0, stream>>>(x, x_mark, y_mark, W_emb, b_emb, scale_p, enc_in, emb_y);
  encoder_gru<<<1, 512, 0, stream>>>(enc_in, W_ih, W_hh, b_ih, b_hh, hT);
  prep_weights<<<3072, 256, 0, stream>>>(W_hh, W_ih, Wbf, w0);
  prep_ck<<<(PRED*G3 + 255)/256, 256, 0, stream>>>(emb_y, W_ih, b_ih, Ck);
  init_state<<<(S_SAMPLES*HIDDEN + 255)/256, 256, 0, stream>>>(hT, x, hA, hbfA, tgt);

  float* hc = hA; float* hn = hB;
  __bf16* hbc = hbfA; __bf16* hbn = hbfB;
  for (int k = 0; k < PRED; ++k){
    // 256 row groups x 4 col groups = 1024 blocks, 8 waves each
    gru_gates_wmma<<<1024, 256, 0, stream>>>(hbc, hc, hn, hbn, Wbf, b_hh, w0, Ck, tgt, k);
    proj_sample<<<(S_SAMPLES + 255)/256, 256, 0, stream>>>(hn, W_df, b_df, W_loc, b_loc,
                                                           W_sc, b_sc, scale_p, tgt, out, k);
    float* tf = hc; hc = hn; hn = tf;
    __bf16* tb = hbc; hbc = hbn; hbn = tb;
  }
  (void)in_sizes; (void)n_in; (void)out_size; (void)ws_size;
}
